// Retriever_32366873542815
// MI455X (gfx1250) — matrix operand order repaired and checked by reading, not compiled, not run on hardware
//
#include <hip/hip_runtime.h>
#include <hip/hip_bf16.h>
#include <stdint.h>

// ---------------------------------------------------------------------------
// Types for WMMA bf16 path (gfx1250, wave32)
// ---------------------------------------------------------------------------
typedef __bf16 bf16_t;
typedef bf16_t v16bf  __attribute__((ext_vector_type(16)));
typedef bf16_t bf16x8 __attribute__((ext_vector_type(8)));
typedef float  v8f    __attribute__((ext_vector_type(8)));

__device__ __forceinline__ bf16_t f2bf(float f) {
  union { float f; uint32_t u; } x; x.f = f;
  uint32_t r = x.u + 0x7FFFu + ((x.u >> 16) & 1u);   // round-to-nearest-even
  union { uint16_t s; bf16_t b; } y; y.s = (uint16_t)(r >> 16);
  return y.b;
}

// ---------------------------------------------------------------------------
// 1) Degree counting for scatter-mean (fwd: dst=t, rev: dst=h)
// ---------------------------------------------------------------------------
__global__ void deg_count(const int* __restrict__ h, const int* __restrict__ t,
                          float* __restrict__ degF, float* __restrict__ degR, int E) {
  int e = blockIdx.x * blockDim.x + threadIdx.x;
  if (e >= E) return;
  atomicAdd(&degF[t[e]], 1.0f);
  atomicAdd(&degR[h[e]], 1.0f);
}

// 2) scatter-add of a [N,2] feature along edges
__global__ void scatter_add2(const float* __restrict__ x, const int* __restrict__ src,
                             const int* __restrict__ dst, float* __restrict__ sum, int E) {
  int e = blockIdx.x * blockDim.x + threadIdx.x;
  if (e >= E) return;
  int s = src[e], d = dst[e];
  atomicAdd(&sum[2 * d + 0], x[2 * s + 0]);
  atomicAdd(&sum[2 * d + 1], x[2 * s + 1]);
}

// 3) divide by max(deg,1) in place
__global__ void div_deg(float* __restrict__ buf, const float* __restrict__ deg, int N) {
  int i = blockIdx.x * blockDim.x + threadIdx.x;
  if (i >= N) return;
  float d = fmaxf(deg[i], 1.0f);
  buf[2 * i + 0] /= d;
  buf[2 * i + 1] /= d;
}

// ---------------------------------------------------------------------------
// 4) Assemble h_e as bf16, K padded 266 -> 288 (9 k-tiles of 32)
// ---------------------------------------------------------------------------
__global__ void build_he(bf16_t* __restrict__ he,
                         const float* __restrict__ entity, const float* __restrict__ nonText,
                         const float* __restrict__ topic,
                         const float* __restrict__ f1, const float* __restrict__ f2,
                         const float* __restrict__ r1, const float* __restrict__ r2,
                         int N, int NT, int total) {
  int idx = blockIdx.x * blockDim.x + threadIdx.x;
  if (idx >= total) return;
  int m = idx / 288, c = idx - m * 288;
  float v = 0.0f;
  if (m < N) {
    if      (c < 256) v = (m < NT) ? entity[(size_t)m * 256 + c] : nonText[c];
    else if (c < 258) v = topic[2 * m + (c - 256)];
    else if (c < 260) v = f1[2 * m + (c - 258)];
    else if (c < 262) v = f2[2 * m + (c - 260)];
    else if (c < 264) v = r1[2 * m + (c - 262)];
    else if (c < 266) v = r2[2 * m + (c - 264)];
  }
  he[idx] = f2bf(v);
}

// 5) relation matrix -> bf16, rows padded R -> Rpad
__global__ void build_rel(bf16_t* __restrict__ dst, const float* __restrict__ rel,
                          int R, int total) {
  int idx = blockIdx.x * blockDim.x + threadIdx.x;
  if (idx >= total) return;
  int m = idx >> 8;
  dst[idx] = (m < R) ? f2bf(rel[idx]) : f2bf(0.0f);
}

// ---------------------------------------------------------------------------
// 6) Pack W1 slices into the WMMA B-tile per-lane layout:
//    dst[((nt*Ktiles + kt)*32 + lane)*16 + i] = W1[rowOff + kt*32 + (i<8 ? half*8+i : 16+half*8+i-8)][col]
// ---------------------------------------------------------------------------
__global__ void pack_w(const float* __restrict__ W1, bf16_t* __restrict__ dst,
                       int Ntiles, int Ktiles, int Kact, int rowOff0, int rowOff1) {
  int idx = blockIdx.x * blockDim.x + threadIdx.x;
  int total = Ntiles * Ktiles * 32;
  if (idx >= total) return;
  int lane = idx & 31;
  int kt   = (idx >> 5) % Ktiles;
  int nt   = idx / (32 * Ktiles);
  int half = lane >> 4;
  int gc   = nt * 16 + (lane & 15);
  int rowOff = (gc < 256) ? rowOff0 : rowOff1;
  int col    = gc & 255;
  bf16_t* d = dst + (size_t)idx * 16;
#pragma unroll
  for (int i = 0; i < 16; ++i) {
    int kk = kt * 32 + ((i < 8) ? (half * 8 + i) : (16 + half * 8 + (i - 8)));
    float v = (kk < Kact) ? W1[(size_t)(rowOff + kk) * 256 + col] : 0.0f;
    d[i] = f2bf(v);
  }
}

// 7) c0 = b1 + q @ W1[0:256,:]
__global__ void compute_c0(const float* __restrict__ q, const float* __restrict__ W1,
                           const float* __restrict__ b1, float* __restrict__ c0) {
  int j = threadIdx.x;  // 256 threads
  float s = b1[j];
  for (int k = 0; k < 256; ++k) s = fmaf(q[k], W1[(size_t)k * 256 + j], s);
  c0[j] = s;
}

// ---------------------------------------------------------------------------
// 8) Register-blocked WMMA GEMM: each wave computes a 32x64 output block
//    (2 m-tiles x 4 n-tiles), reusing A fragments 4x and B fragments 2x:
//    operand traffic drops from 2048 B/WMMA to 768 B/WMMA.
//    A row-major bf16 (lda elems); B pre-packed per-lane-contiguous.
// ---------------------------------------------------------------------------
__global__ void wmma_gemm_2x4(const bf16_t* __restrict__ A, int lda,
                              const bf16_t* __restrict__ Wpack,
                              float* __restrict__ out, int ldo,
                              int Mb, int Nb, int Ktiles) {
  int wave = blockIdx.x * (blockDim.x >> 5) + (threadIdx.x >> 5);
  int lane = threadIdx.x & 31;
  if (wave >= Mb * Nb) return;           // uniform per wave -> EXEC all-ones inside
  int mb = wave / Nb;
  int nb = wave - mb * Nb;
  int half = lane >> 4;

  const bf16_t* arow0 = A + (size_t)(mb * 32 + (lane & 15)) * lda + half * 8;
  const bf16_t* arow1 = arow0 + (size_t)16 * lda;
  const bf16_t* bbase = Wpack + ((size_t)(nb * 4) * Ktiles * 32 + lane) * 16;

  v8f acc[2][4] = {};
  for (int kt = 0; kt < Ktiles; ++kt) {
    v16bf a[2], b[4];
    {
      bf16x8 lo0 = *(const bf16x8*)(arow0 + kt * 32);
      bf16x8 hi0 = *(const bf16x8*)(arow0 + kt * 32 + 16);
      bf16x8 lo1 = *(const bf16x8*)(arow1 + kt * 32);
      bf16x8 hi1 = *(const bf16x8*)(arow1 + kt * 32 + 16);
#pragma unroll
      for (int i = 0; i < 8; ++i) {
        a[0][i] = lo0[i]; a[0][i + 8] = hi0[i];
        a[1][i] = lo1[i]; a[1][i + 8] = hi1[i];
      }
    }
#pragma unroll
    for (int ni = 0; ni < 4; ++ni)
      b[ni] = *(const v16bf*)(bbase + ((size_t)ni * Ktiles + kt) * 32 * 16);
#pragma unroll
    for (int mi = 0; mi < 2; ++mi)
#pragma unroll
      for (int ni = 0; ni < 4; ++ni)
        acc[mi][ni] = __builtin_amdgcn_wmma_f32_16x16x32_bf16(
            false, a[mi], false, b[ni], (short)0, acc[mi][ni], false, false);
  }

#pragma unroll
  for (int mi = 0; mi < 2; ++mi)
#pragma unroll
    for (int ni = 0; ni < 4; ++ni) {
      float* orow = out + (size_t)(mb * 32 + mi * 16 + (half << 3)) * ldo +
                    (nb * 4 + ni) * 16 + (lane & 15);
#pragma unroll
      for (int v = 0; v < 8; ++v) orow[(size_t)v * ldo] = acc[mi][ni][v];
    }
}

// ---------------------------------------------------------------------------
// 9) Per-edge fused MLP: one wave per edge, fully coalesced float4 gathers.
//    hidden = relu(c0 + Pht[h,0:256] + relP[r] + Pht[t,256:512]); out = hidden.W2 + b2
// ---------------------------------------------------------------------------
__global__ void edge_mlp(const int* __restrict__ h_id, const int* __restrict__ r_id,
                         const int* __restrict__ t_id,
                         const float* __restrict__ Pht, const float* __restrict__ relP,
                         const float* __restrict__ c0, const float* __restrict__ W2,
                         const float* __restrict__ b2, float* __restrict__ out, int E) {
  int e = blockIdx.x * (blockDim.x >> 5) + (threadIdx.x >> 5);
  int lane = threadIdx.x & 31;
  if (e >= E) return;  // uniform per wave
  const float4* Hp  = (const float4*)(Pht + (size_t)h_id[e] * 512);
  const float4* Tp  = (const float4*)(Pht + (size_t)t_id[e] * 512 + 256);
  const float4* Rp  = (const float4*)(relP + (size_t)r_id[e] * 256);
  const float4* C0  = (const float4*)c0;
  const float4* W2v = (const float4*)W2;
  float acc = 0.0f;
#pragma unroll
  for (int qq = 0; qq < 2; ++qq) {
    int j = qq * 32 + lane;  // lanes contiguous -> 512 B coalesced per stream
    float4 hv = Hp[j], tv = Tp[j], rv = Rp[j], cv = C0[j], wv = W2v[j];
    acc += fmaxf(cv.x + hv.x + rv.x + tv.x, 0.0f) * wv.x;
    acc += fmaxf(cv.y + hv.y + rv.y + tv.y, 0.0f) * wv.y;
    acc += fmaxf(cv.z + hv.z + rv.z + tv.z, 0.0f) * wv.z;
    acc += fmaxf(cv.w + hv.w + rv.w + tv.w, 0.0f) * wv.w;
  }
  for (int off = 16; off > 0; off >>= 1) acc += __shfl_down(acc, off);
  if (lane == 0) out[e] = acc + b2[0];
}

// ---------------------------------------------------------------------------
// Host-side orchestration
// ---------------------------------------------------------------------------
extern "C" void kernel_launch(void* const* d_in, const int* in_sizes, int n_in,
                              void* d_out, int out_size, void* d_ws, size_t ws_size,
                              hipStream_t stream) {
  const int*   h_id     = (const int*)  d_in[0];
  const int*   r_id     = (const int*)  d_in[1];
  const int*   t_id     = (const int*)  d_in[2];
  const float* q_emb    = (const float*)d_in[3];
  const float* entity   = (const float*)d_in[4];
  const float* rel_embs = (const float*)d_in[6];
  const float* topic    = (const float*)d_in[7];
  const float* non_text = (const float*)d_in[8];
  const float* W1       = (const float*)d_in[9];
  const float* b1       = (const float*)d_in[10];
  const float* W2       = (const float*)d_in[11];
  const float* b2       = (const float*)d_in[12];

  const int E  = in_sizes[0];
  const int N  = in_sizes[7] / 2;      // 110000
  const int NT = in_sizes[4] / 256;    // 100000
  const int R  = in_sizes[6] / 256;    // 500

  const int Npad   = (N + 31) & ~31;   // pad M to 32 rows for 2-wide m-blocking
  const int Rpad   = (R + 31) & ~31;   // 512
  const int KheT   = 9;                // 288 / 32  (266 padded to 288)
  const int KrelT  = 8;                // 256 / 32
  const int NtHT   = 32;               // 512 output cols = [W1_h | W1_t]
  const int NtR    = 16;               // 256 output cols

  // ---- workspace carve (256 B aligned) ----
  char* base = (char*)d_ws;
  size_t off = 0;
  auto carve = [&](size_t bytes) -> char* {
    char* p = base + off;
    off = (off + bytes + 255) & ~(size_t)255;
    return p;
  };
  char* zeroBase = base + off;
  float* degF = (float*)carve((size_t)N * 4);
  float* degR = (float*)carve((size_t)N * 4);
  float* f1   = (float*)carve((size_t)N * 8);
  float* f2   = (float*)carve((size_t)N * 8);
  float* r1   = (float*)carve((size_t)N * 8);
  float* r2   = (float*)carve((size_t)N * 8);
  size_t zeroBytes = (size_t)((base + off) - zeroBase);

  float*  c0     = (float*) carve(256 * 4);
  bf16_t* WpkHT  = (bf16_t*)carve((size_t)NtHT * KheT * 32 * 16 * 2);
  bf16_t* WpkR   = (bf16_t*)carve((size_t)NtR * KrelT * 32 * 16 * 2);
  bf16_t* relBf  = (bf16_t*)carve((size_t)Rpad * 256 * 2);
  float*  relP   = (float*) carve((size_t)Rpad * 256 * 4);
  bf16_t* heBf   = (bf16_t*)carve((size_t)Npad * 288 * 2);
  float*  Pht    = (float*) carve((size_t)Npad * 512 * 4);
  (void)ws_size; // total ~295 MB

  float* out = (float*)d_out;

  const int TB = 256;
  dim3 blk(TB);

  // ---- DDE scatter-mean rounds ----
  hipMemsetAsync(zeroBase, 0, zeroBytes, stream);
  deg_count<<<dim3((E + TB - 1) / TB), blk, 0, stream>>>(h_id, t_id, degF, degR, E);

  scatter_add2<<<dim3((E + TB - 1) / TB), blk, 0, stream>>>(topic, h_id, t_id, f1, E);
  div_deg<<<dim3((N + TB - 1) / TB), blk, 0, stream>>>(f1, degF, N);
  scatter_add2<<<dim3((E + TB - 1) / TB), blk, 0, stream>>>(f1, h_id, t_id, f2, E);
  div_deg<<<dim3((N + TB - 1) / TB), blk, 0, stream>>>(f2, degF, N);
  scatter_add2<<<dim3((E + TB - 1) / TB), blk, 0, stream>>>(topic, t_id, h_id, r1, E);
  div_deg<<<dim3((N + TB - 1) / TB), blk, 0, stream>>>(r1, degR, N);
  scatter_add2<<<dim3((E + TB - 1) / TB), blk, 0, stream>>>(r1, t_id, h_id, r2, E);
  div_deg<<<dim3((N + TB - 1) / TB), blk, 0, stream>>>(r2, degR, N);

  // ---- operand staging ----
  {
    int total = Npad * 288;
    build_he<<<dim3((total + TB - 1) / TB), blk, 0, stream>>>(
        heBf, entity, non_text, topic, f1, f2, r1, r2, N, NT, total);
  }
  {
    int total = Rpad * 256;
    build_rel<<<dim3((total + TB - 1) / TB), blk, 0, stream>>>(relBf, rel_embs, R, total);
  }
  pack_w<<<dim3((NtHT * KheT * 32 + TB - 1) / TB), blk, 0, stream>>>(
      W1, WpkHT, NtHT, KheT, 266, /*W1_h*/ 256, /*W1_t*/ 778);
  pack_w<<<dim3((NtR * KrelT * 32 + TB - 1) / TB), blk, 0, stream>>>(
      W1, WpkR, NtR, KrelT, 256, /*W1_r*/ 522, 522);
  compute_c0<<<dim3(1), blk, 0, stream>>>(q_emb, W1, b1, c0);

  // ---- WMMA projection GEMMs (2x4 register-blocked) ----
  {
    int Mb = Npad / 32, Nb = NtHT / 4;     // 3438 * 8 waves
    int waves = Mb * Nb;
    wmma_gemm_2x4<<<dim3((waves + 7) / 8), blk, 0, stream>>>(
        heBf, 288, WpkHT, Pht, 512, Mb, Nb, KheT);
  }
  {
    int Mb = Rpad / 32, Nb = NtR / 4;      // 16 * 4 waves
    int waves = Mb * Nb;
    wmma_gemm_2x4<<<dim3((waves + 7) / 8), blk, 0, stream>>>(
        relBf, 256, WpkR, relP, 256, Mb, Nb, KrelT);
  }

  // ---- fused per-edge MLP (one wave per edge) ----
  edge_mlp<<<dim3((E + 7) / 8), blk, 0, stream>>>(
      h_id, r_id, t_id, Pht, relP, c0, W2, b2, out, E);
  (void)out_size; (void)n_in;
}
